// Encoder_20169166422855
// MI455X (gfx1250) — matrix operand order repaired and checked by reading, not compiled, not run on hardware
//
#include <hip/hip_runtime.h>
#include <hip/hip_bf16.h>
#include <math.h>

typedef unsigned short u16;
typedef unsigned int   u32;
typedef __attribute__((ext_vector_type(16))) __bf16 v16bf;
typedef __attribute__((ext_vector_type(8)))  float  v8f;

#define BB_    2
#define DIM_   256
#define HEADS_ 8
#define WVOL_  256
#define NWIN_  98            // windows per batch (2*7*7)
#define NWB_   (BB_*NWIN_)   // 196
#define MTOK_  (NWB_*WVOL_)  // 50176 tokens
#define MLP_   1024
#define NBIAS_ 1575

// ---------- helpers ----------
__device__ __forceinline__ u16 f2bf(float f){
  u32 u = __float_as_uint(f);
  u32 r = u + 0x7fffu + ((u >> 16) & 1u);   // round-to-nearest-even
  return (u16)(r >> 16);
}

union BfVec { v16bf v; uint4 q[2]; };

// A-matrix 16x32 bf16, row-major source (lda in elements).
// lane<16 holds row=lane, K=0..15; lane>=16 holds row=lane-16, K=16..31
__device__ __forceinline__ v16bf load_A16x32(const u16* src, int lda, int lane){
  int m  = lane & 15;
  int kb = (lane < 16) ? 0 : 16;
  const u16* p = src + m*lda + kb;
  BfVec t;
  t.q[0] = *(const uint4*)(p);
  t.q[1] = *(const uint4*)(p + 8);
  return t.v;
}
// B-matrix 32x16 bf16 from transposed source: srcT[n][k], ldk elements per row.
// lane<16: col=lane, K=0..15; lane>=16: col=lane-16, K=16..31 (contiguous 32B per lane)
__device__ __forceinline__ v16bf load_B32x16(const u16* srcT, int ldk, int lane){
  int n  = lane & 15;
  int kb = (lane < 16) ? 0 : 16;
  const u16* p = srcT + n*ldk + kb;
  BfVec t;
  t.q[0] = *(const uint4*)(p);
  t.q[1] = *(const uint4*)(p + 8);
  return t.v;
}
__device__ __forceinline__ v8f wmma_bf16(v16bf a, v16bf b, v8f c){
  return __builtin_amdgcn_wmma_f32_16x16x32_bf16(false, a, false, b, (short)0, c, false, false);
}
__device__ __forceinline__ void vzero(v8f& v){
#pragma unroll
  for(int i=0;i<8;i++) v[i] = 0.0f;
}

// async global->LDS copy, 16B per lane (ASYNCcnt-tracked, CDNA5)
__device__ __forceinline__ void async_g2l_b128(void* lds_dst, const void* gsrc){
  u32 loff = (u32)(uintptr_t)lds_dst;   // flat shared ptr: low 32 bits = LDS byte address
  asm volatile("global_load_async_to_lds_b128 %0, %1, off"
               :: "v"(loff), "v"(gsrc) : "memory");
}

// fully-unrolled ping-pong 16x64 wave-tile GEMM (bf16 x bf16 -> f32).
// KTOT must be a multiple of 64. Full unroll removes all loop-carried vector
// registers (no rotation movs); the ping-pong ordering keeps the next
// 32-chunk's loads in flight across each wmma burst.
template<int KTOT>
__device__ __forceinline__ void gemm_tile_16x64(const u16* __restrict__ Ar, int lda,
                                                const u16* __restrict__ BT, int ldb,
                                                int lane, v8f acc[4]){
  v16bf a0 = load_A16x32(Ar, lda, lane);
  v16bf b0[4];
#pragma unroll
  for(int t=0;t<4;t++) b0[t] = load_B32x16(BT + t*16*ldb, ldb, lane);
  v16bf a1; v16bf b1[4];
#pragma unroll
  for(int k=0; k<KTOT-64; k+=64){
    __builtin_prefetch((const void*)(Ar + k + 128), 0, 3);
    a1 = load_A16x32(Ar + k + 32, lda, lane);
#pragma unroll
    for(int t=0;t<4;t++) b1[t] = load_B32x16(BT + t*16*ldb + k + 32, ldb, lane);
#pragma unroll
    for(int t=0;t<4;t++) acc[t] = wmma_bf16(a0, b0[t], acc[t]);
    a0 = load_A16x32(Ar + k + 64, lda, lane);
#pragma unroll
    for(int t=0;t<4;t++) b0[t] = load_B32x16(BT + t*16*ldb + k + 64, ldb, lane);
#pragma unroll
    for(int t=0;t<4;t++) acc[t] = wmma_bf16(a1, b1[t], acc[t]);
  }
  // tail 64-chunk: k = KTOT-64
  a1 = load_A16x32(Ar + KTOT - 32, lda, lane);
#pragma unroll
  for(int t=0;t<4;t++) b1[t] = load_B32x16(BT + t*16*ldb + KTOT - 32, ldb, lane);
#pragma unroll
  for(int t=0;t<4;t++) acc[t] = wmma_bf16(a0, b0[t], acc[t]);
#pragma unroll
  for(int t=0;t<4;t++) acc[t] = wmma_bf16(a1, b1[t], acc[t]);
}

// ---------- one-time prep ----------
__global__ void wprep_kernel(const float* __restrict__ src, u16* __restrict__ dst, int K, int N){
  int i = blockIdx.x*256 + threadIdx.x;
  if(i >= K*N) return;
  int k = i / N, n = i % N;
  dst[n*K + k] = f2bf(src[i]);
}

__global__ void bias_expand_kernel(const float* __restrict__ tab, float* __restrict__ out){
  int i = blockIdx.x*256 + threadIdx.x;        // 4*8*256*256 = 2^21 total
  int j  = i & 255;
  int qi = (i >> 8) & 255;
  int h  = (i >> 16) & 7;
  int l  = i >> 19;
  int pdi = qi >> 6, phi = (qi >> 3) & 7, pwi = qi & 7;
  int pdj = j  >> 6, phj = (j  >> 3) & 7, pwj = j  & 7;
  int idx = (pdi - pdj + 3)*225 + (phi - phj + 7)*15 + (pwi - pwj + 7);
  out[i] = tab[(l*NBIAS_ + idx)*8 + h];
}

// ---------- patch-merge conv (1,4,4)/(1,4,4) -> y [tok][256] f32 ----------
__global__ void conv_patch_kernel(const float* __restrict__ x, const float* __restrict__ w,
                                  const float* __restrict__ bias, float* __restrict__ y){
  long i = (long)blockIdx.x*256 + threadIdx.x;
  if(i >= (long)MTOK_*DIM_) return;
  int c  = (int)(i & 255);
  long tok = i >> 8;
  int wp = (int)(tok % 56);
  int hp = (int)((tok/56) % 56);
  int d  = (int)((tok/3136) & 7);
  int b  = (int)(tok / 25088);
  float acc = bias[c];
#pragma unroll
  for(int ic=0; ic<4; ic++){
    const float* xp = x + ((((long)b*4 + ic)*8 + d)*224 + hp*4)*224 + wp*4;
    const float* wp2 = w + (c*4 + ic)*16;
#pragma unroll
    for(int kh=0; kh<4; kh++)
#pragma unroll
      for(int kw=0; kw<4; kw++)
        acc += xp[kh*224 + kw] * wp2[kh*4 + kw];
  }
  y[i] = acc;
}

// ---------- layernorm (one wave per token), optional windowed/shifted gather ----------
__global__ __launch_bounds__(256) void ln_kernel(const float* __restrict__ y,
        const float* __restrict__ s, const float* __restrict__ bsh,
        u16* __restrict__ dst, int windowed, int shifted){
  int lane = threadIdx.x & 31;
  int tok  = blockIdx.x*8 + (threadIdx.x >> 5);
  const float* row = y + (long)tok*256;
  float v[8];
  float sum = 0.f, sq = 0.f;
#pragma unroll
  for(int e=0;e<8;e++){ v[e] = row[lane + 32*e]; sum += v[e]; sq += v[e]*v[e]; }
#pragma unroll
  for(int mm=1; mm<32; mm<<=1){ sum += __shfl_xor(sum, mm, 32); sq += __shfl_xor(sq, mm, 32); }
  float mean = sum*(1.f/256.f);
  float var  = sq*(1.f/256.f) - mean*mean;
  float rstd = rsqrtf(var + 1e-5f);
  long drow;
  if(windowed){
    int wp = tok % 56, hp = (tok/56) % 56, d = (tok/3136) & 7, b = tok/25088;
    int gd = d, gh = hp, gw = wp;
    if(shifted){ gd = (d + 6) & 7; gh = (hp + 52) % 56; gw = (wp + 52) % 56; }
    int win = (gd >> 2)*49 + (gh >> 3)*7 + (gw >> 3);
    int pos = (gd & 3)*64 + (gh & 7)*8 + (gw & 7);
    drow = (long)(b*NWIN_ + win)*256 + pos;
  } else drow = tok;
  u16* out = dst + drow*256;
#pragma unroll
  for(int e=0;e<8;e++){
    int c = lane + 32*e;
    out[c] = f2bf((v[e]-mean)*rstd*s[c] + bsh[c]);
  }
}

// ---------- QKV GEMM: Xw[50176x256] @ WT[768x256]^T -> Q,K row-major, V transposed ----------
__global__ __launch_bounds__(256) void qkv_gemm_kernel(const u16* __restrict__ A,
      const u16* __restrict__ WT, u16* __restrict__ Qb, u16* __restrict__ Kb, u16* __restrict__ Vt){
  int wid  = (blockIdx.x*256 + threadIdx.x) >> 5;
  int lane = threadIdx.x & 31;
  int mt = wid / 12, ntg = wid % 12;
  int mbase = mt*16, nbase = ntg*64;
  v8f acc[4];
#pragma unroll
  for(int t=0;t<4;t++) vzero(acc[t]);
  gemm_tile_16x64<256>(A + (long)mbase*256, 256, WT + nbase*256, 256, lane, acc);
  int nl = lane & 15, hof = (lane < 16) ? 0 : 8;
#pragma unroll
  for(int t=0;t<4;t++){
    int n = nbase + t*16 + nl;
    int part = n >> 8, head = (n >> 5) & 7, hd = n & 31;
#pragma unroll
    for(int r=0;r<8;r++){
      int m = mbase + r + hof;
      int win = m >> 8, tokp = m & 255;
      float val = acc[t][r];
      if(part == 0)      Qb[(((win<<3)+head)*256 + tokp)*32 + hd] = f2bf(val * 0.17677669529663687f);
      else if(part == 1) Kb[(((win<<3)+head)*256 + tokp)*32 + hd] = f2bf(val);
      else               Vt[(((win<<3)+head)*32 + hd)*256 + tokp] = f2bf(val);
    }
  }
}

// ---------- flash-style windowed attention: 1 workgroup per (window,head), 16 waves x 16 rows ----------
__global__ __launch_bounds__(512) void attn_kernel(const u16* __restrict__ Qb, const u16* __restrict__ Kb,
      const u16* __restrict__ Vt, const float* __restrict__ bias_l, u16* __restrict__ Oout, int shifted){
  __shared__ __align__(16) u16 sK[8192];   // [tok][32]
  __shared__ __align__(16) u16 sV[8192];   // [hd][tok]
  __shared__ __align__(16) u16 sP[8192];   // per-wave 16x32 staging
  __shared__ int sGid[256];
  int wh = blockIdx.x;
  int win = wh >> 3, head = wh & 7;
  int tid = threadIdx.x, lane = tid & 31, wave = tid >> 5;
  {
    // async global->LDS staging of K and V^T panels (ASYNCcnt path)
    const uint4* gK = (const uint4*)(Kb + (long)wh*8192);
    const uint4* gV = (const uint4*)(Vt + (long)wh*8192);
    uint4* lK = (uint4*)sK; uint4* lV = (uint4*)sV;
    for(int i=tid; i<1024; i+=512){
      async_g2l_b128(lK + i, gK + i);
      async_g2l_b128(lV + i, gV + i);
    }
    asm volatile("s_wait_asynccnt 0x0" ::: "memory");
  }
  if(tid < 256){
    int w = win % NWIN_;
    int wd = w/49, whh = (w/7)%7, ww = w%7;
    int gd = wd*4 + (tid >> 6), gh = whh*8 + ((tid >> 3) & 7), gw = ww*8 + (tid & 7);
    int rd = gd < 4 ? 0 : (gd < 6 ? 1 : 2);
    int rh = gh < 48 ? 0 : (gh < 52 ? 1 : 2);
    int rw = gw < 48 ? 0 : (gw < 52 ? 1 : 2);
    sGid[tid] = rd*9 + rh*3 + rw;
  }
  __syncthreads();

  int qbase = wave*16;
  int nl = lane & 15, hof = (lane < 16) ? 0 : 8;
  v16bf aq = load_A16x32(Qb + (long)wh*8192 + qbase*32, 32, lane);
  float mrun[8], lrun[8];
  v8f o0, o1; vzero(o0); vzero(o1);
#pragma unroll
  for(int r=0;r<8;r++){ mrun[r] = -3.0e38f; lrun[r] = 0.f; }
  const float* bprow = bias_l + head*65536;
  u16* pP = sP + wave*512;

  for(int kc=0; kc<8; kc++){
    int jb = kc*32;
    v16bf b0 = load_B32x16(sK + jb*32, 32, lane);
    v16bf b1 = load_B32x16(sK + (jb+16)*32, 32, lane);
    v8f s0, s1; vzero(s0); vzero(s1);
    s0 = wmma_bf16(aq, b0, s0);               // S tile cols jb..jb+15
    s1 = wmma_bf16(aq, b1, s1);               // S tile cols jb+16..jb+31
    int gk0 = 0, gk1 = 0;
    if(shifted){ gk0 = sGid[jb + nl]; gk1 = sGid[jb + 16 + nl]; }
#pragma unroll
    for(int r=0;r<8;r++){
      int i = qbase + r + hof;                // query row (C/D layout)
      s0[r] += bprow[i*256 + jb + nl];
      s1[r] += bprow[i*256 + jb + 16 + nl];
      if(shifted){
        int gq = sGid[i];
        if(gq != gk0) s0[r] -= 1e9f;
        if(gq != gk1) s1[r] -= 1e9f;
      }
      float cm = fmaxf(s0[r], s1[r]);
#pragma unroll
      for(int mm=1; mm<16; mm<<=1) cm = fmaxf(cm, __shfl_xor(cm, mm, 32));
      float nm = fmaxf(mrun[r], cm);
      float sc = __expf(mrun[r] - nm);
      mrun[r] = nm;
      float p0 = __expf(s0[r] - nm), p1 = __expf(s1[r] - nm);
      float rs = p0 + p1;
#pragma unroll
      for(int mm=1; mm<16; mm<<=1) rs += __shfl_xor(rs, mm, 32);
      lrun[r] = lrun[r]*sc + rs;
      o0[r] *= sc; o1[r] *= sc;
      int il = r + hof;
      pP[il*32 + nl]      = f2bf(p0);         // stage P row-major 16x32
      pP[il*32 + 16 + nl] = f2bf(p1);
    }
    asm volatile("s_wait_storecnt_dscnt 0x0" ::: "memory");  // cross-lane LDS RAW inside the wave
    v16bf ap  = load_A16x32(pP, 32, lane);
    v16bf bv0 = load_B32x16(sV + jb,           256, lane);   // hd 0..15
    v16bf bv1 = load_B32x16(sV + 16*256 + jb,  256, lane);   // hd 16..31
    o0 = wmma_bf16(ap, bv0, o0);
    o1 = wmma_bf16(ap, bv1, o1);
  }
#pragma unroll
  for(int r=0;r<8;r++){
    int i = qbase + r + hof;
    float inv = 1.f / lrun[r];
    long rowb = ((long)win*256 + i)*256 + head*32;
    Oout[rowb + nl]      = f2bf(o0[r]*inv);
    Oout[rowb + 16 + nl] = f2bf(o1[r]*inv);
  }
}

// ---------- out-proj GEMM + bias + residual scatter (undo windowing/shift) ----------
__global__ __launch_bounds__(256) void outproj_gemm_kernel(const u16* __restrict__ A,
      const u16* __restrict__ WT, const float* __restrict__ bias, float* __restrict__ y, int shifted){
  int wid  = (blockIdx.x*256 + threadIdx.x) >> 5;
  int lane = threadIdx.x & 31;
  int mt = wid >> 2, ntg = wid & 3;
  int mbase = mt*16, nbase = ntg*64;
  v8f acc[4];
#pragma unroll
  for(int t=0;t<4;t++) vzero(acc[t]);
  gemm_tile_16x64<256>(A + (long)mbase*256, 256, WT + nbase*256, 256, lane, acc);
  int nl = lane & 15, hof = (lane < 16) ? 0 : 8;
#pragma unroll
  for(int t=0;t<4;t++){
    int n = nbase + t*16 + nl;
    float bv = bias[n];
#pragma unroll
    for(int r=0;r<8;r++){
      int m = mbase + r + hof;
      int win = m >> 8, pos = m & 255;
      int b = win / NWIN_, w = win % NWIN_;
      int wd = w/49, whh = (w/7)%7, ww = w%7;
      int gd = wd*4 + (pos >> 6), gh = whh*8 + ((pos >> 3) & 7), gw = ww*8 + (pos & 7);
      if(shifted){ gd = (gd + 2) & 7; gh = (gh + 4) % 56; gw = (gw + 4) % 56; }
      long ti = ((long)(b*8 + gd)*56 + gh)*56 + gw;
      y[ti*256 + n] += acc[t][r] + bv;
    }
  }
}

// ---------- MLP GEMMs ----------
__global__ __launch_bounds__(256) void fc1_gemm_kernel(const u16* __restrict__ A,
      const u16* __restrict__ WT, const float* __restrict__ bias, u16* __restrict__ H){
  int wid  = (blockIdx.x*256 + threadIdx.x) >> 5;
  int lane = threadIdx.x & 31;
  int mt = wid >> 4, ntg = wid & 15;
  int mbase = mt*16, nbase = ntg*64;
  v8f acc[4];
#pragma unroll
  for(int t=0;t<4;t++) vzero(acc[t]);
  gemm_tile_16x64<256>(A + (long)mbase*256, 256, WT + nbase*256, 256, lane, acc);
  int nl = lane & 15, hof = (lane < 16) ? 0 : 8;
#pragma unroll
  for(int t=0;t<4;t++){
    int n = nbase + t*16 + nl;
    float bv = bias[n];
#pragma unroll
    for(int r=0;r<8;r++){
      int m = mbase + r + hof;
      float xg = acc[t][r] + bv;
      float g  = 0.5f*xg*(1.f + tanhf(0.7978845608f*(xg + 0.044715f*xg*xg*xg)));
      H[(long)m*MLP_ + n] = f2bf(g);
    }
  }
}

__global__ __launch_bounds__(256) void fc2_gemm_kernel(const u16* __restrict__ A,
      const u16* __restrict__ WT, const float* __restrict__ bias, float* __restrict__ y){
  int wid  = (blockIdx.x*256 + threadIdx.x) >> 5;
  int lane = threadIdx.x & 31;
  int mt = wid >> 2, ntg = wid & 3;
  int mbase = mt*16, nbase = ntg*64;
  v8f acc[4];
#pragma unroll
  for(int t=0;t<4;t++) vzero(acc[t]);
  gemm_tile_16x64<1024>(A + (long)mbase*MLP_, MLP_, WT + nbase*MLP_, MLP_, lane, acc);
  int nl = lane & 15, hof = (lane < 16) ? 0 : 8;
#pragma unroll
  for(int t=0;t<4;t++){
    int n = nbase + t*16 + nl;
    float bv = bias[n];
#pragma unroll
    for(int r=0;r<8;r++){
      int m = mbase + r + hof;
      y[(long)m*256 + n] += acc[t][r] + bv;
    }
  }
}

// ---------- final channel-first transpose ----------
__global__ void out_transpose_kernel(const float* __restrict__ y, float* __restrict__ out){
  long i = (long)blockIdx.x*256 + threadIdx.x;
  if(i >= (long)MTOK_*DIM_) return;
  int wp = (int)(i % 56); long r = i/56;
  int hp = (int)(r % 56); r /= 56;
  int d  = (int)(r & 7);  r >>= 3;
  int c  = (int)(r & 255);
  int b  = (int)(r >> 8);
  out[i] = y[(((long)(b*8 + d)*56 + hp)*56 + wp)*256 + c];
}

// ---------- host-side launch ----------
extern "C" void kernel_launch(void* const* d_in, const int* in_sizes, int n_in,
                              void* d_out, int out_size, void* d_ws, size_t ws_size,
                              hipStream_t stream){
  (void)in_sizes; (void)n_in; (void)out_size; (void)ws_size;
  const float* x      = (const float*)d_in[0];
  const float* conv_w = (const float*)d_in[1];
  const float* conv_b = (const float*)d_in[2];
  const float* ln1_s  = (const float*)d_in[3];
  const float* ln1_b  = (const float*)d_in[4];
  const float* qkv_w  = (const float*)d_in[5];
  const float* out_w  = (const float*)d_in[6];
  const float* out_b  = (const float*)d_in[7];
  const float* bias_t = (const float*)d_in[8];
  const float* ln2_s  = (const float*)d_in[9];
  const float* ln2_b  = (const float*)d_in[10];
  const float* fc1_w  = (const float*)d_in[11];
  const float* fc1_b  = (const float*)d_in[12];
  const float* fc2_w  = (const float*)d_in[13];
  const float* fc2_b  = (const float*)d_in[14];

  char* ws = (char*)d_ws;
  float* y     = (float*)(ws + 0);               // 51,380,224 B
  u16*  qkvT   = (u16*)(ws + 51380224);          //  1,572,864 B
  u16*  outT   = (u16*)(ws + 52953088);          //    524,288 B
  u16*  fc1T   = (u16*)(ws + 53477376);          //  2,097,152 B
  u16*  fc2T   = (u16*)(ws + 55574528);          //  2,097,152 B
  float* biasP = (float*)(ws + 57671680);        //  8,388,608 B
  char* blockA = ws + 66060288;                  // 102,760,448 B (Xw|Q|K|Vt aliased with Hbuf)
  u16*  Xw     = (u16*)(blockA);
  u16*  Qb     = (u16*)(blockA + 25690112);
  u16*  Kb     = (u16*)(blockA + 51380224);
  u16*  Vt     = (u16*)(blockA + 77070336);
  u16*  Hbuf   = (u16*)(blockA);                 // alias: live only during MLP
  u16*  Oout   = (u16*)(ws + 168820736);         // 25,690,112 B
  u16*  Xn     = (u16*)(ws + 194510848);         // 25,690,112 B

  for(int l=0; l<4; l++){
    wprep_kernel<<<(256*768  + 255)/256, 256, 0, stream>>>(qkv_w + l*256*768,  qkvT + l*768*256,  256, 768);
    wprep_kernel<<<(256*256  + 255)/256, 256, 0, stream>>>(out_w + l*256*256,  outT + l*256*256,  256, 256);
    wprep_kernel<<<(256*1024 + 255)/256, 256, 0, stream>>>(fc1_w + l*256*1024, fc1T + l*1024*256, 256, 1024);
    wprep_kernel<<<(1024*256 + 255)/256, 256, 0, stream>>>(fc2_w + l*1024*256, fc2T + l*256*1024, 1024, 256);
  }
  bias_expand_kernel<<<8192, 256, 0, stream>>>(bias_t, biasP);
  conv_patch_kernel<<<50176, 256, 0, stream>>>(x, conv_w, conv_b, y);

  for(int l=0; l<4; l++){
    int shifted = l & 1;
    ln_kernel<<<6272, 256, 0, stream>>>(y, ln1_s + l*256, ln1_b + l*256, Xw, 1, shifted);
    qkv_gemm_kernel<<<4704, 256, 0, stream>>>(Xw, qkvT + l*768*256, Qb, Kb, Vt);
    attn_kernel<<<NWB_*HEADS_, 512, 0, stream>>>(Qb, Kb, Vt, biasP + l*8*65536, Oout, shifted);
    outproj_gemm_kernel<<<1568, 256, 0, stream>>>(Oout, outT + l*256*256, out_b + l*256, y, shifted);
    ln_kernel<<<6272, 256, 0, stream>>>(y, ln2_s + l*256, ln2_b + l*256, Xn, 0, 0);
    fc1_gemm_kernel<<<6272, 256, 0, stream>>>(Xn, fc1T + l*1024*256, fc1_b + l*1024, Hbuf);
    fc2_gemm_kernel<<<1568, 256, 0, stream>>>(Hbuf, fc2T + l*256*1024, fc2_b + l*256, y);
  }
  out_transpose_kernel<<<50176, 256, 0, stream>>>(y, (float*)d_out);
}
